// SpatialAttentionLayer_GAT_21311627723297
// MI455X (gfx1250) — matrix operand order repaired and checked by reading, not compiled, not run on hardware
//
#include <hip/hip_runtime.h>
#include <math.h>

// ---------------------------------------------------------------------------
// GATv2 x4 + output projection for MI455X (gfx1250, wave32, WMMA).
// GEMMs use v_wmma_f32_16x16x32_bf16 with a bf16 hi/lo split (bf16x3) so the
// projections keep ~fp32 accuracy. Edge segment-softmax is atomic-based and
// L2-resident (num/denom buffers << 192MB L2).
// ---------------------------------------------------------------------------

typedef __attribute__((ext_vector_type(16))) __bf16 v16bf;
typedef __attribute__((ext_vector_type(8)))  float  v8f;

#define NEG_SLOPE 0.2f

__device__ __forceinline__ void atomic_max_f32(float* addr, float val) {
  // monotonic-bits trick: works for all finite floats, init must be -inf
  if (val >= 0.0f) {
    atomicMax((int*)addr, __float_as_int(val));
  } else {
    atomicMin((unsigned int*)addr, (unsigned int)__float_as_int(val));
  }
}

// ---------------------------------------------------------------------------
// Y[N,64] = X[N,K(ldx)] @ W[K,64]  (+ bias), bf16x3 WMMA.
// block = 256 threads (8 waves): 32 rows x 64 cols, one 16x16 tile per wave.
// ---------------------------------------------------------------------------
__global__ void __launch_bounds__(256)
gemm_wmma_bf16x3(const float* __restrict__ X, int ldx, int K,
                 const float* __restrict__ W,
                 const float* __restrict__ bias,
                 float* __restrict__ Y, int N)
{
  const int lane = threadIdx.x & 31;
  const int wave = threadIdx.x >> 5;
  const int r    = wave >> 2;          // 0..1  (row tile)
  const int c    = wave & 3;           // 0..3  (col tile)
  const int m0   = blockIdx.x * 32 + r * 16;
  const int n0   = c * 16;
  const int half = lane >> 4;          // 0 or 1
  const int l16  = lane & 15;

  const int mrow   = m0 + l16;
  const int mclamp = (mrow < N) ? mrow : (N - 1);
  const float* xrow = X + (size_t)mclamp * ldx;

  v8f acc = {};

  for (int kc = 0; kc < K; kc += 32) {
    // ---- A fragment: 16x32 bf16 (M = l16; K pattern per ISA layout) ----
    float af[16];
    const int offA = half * 8;
#pragma unroll
    for (int j = 0; j < 8; ++j) af[j]     = xrow[kc + offA + j];
#pragma unroll
    for (int j = 0; j < 8; ++j) af[8 + j] = xrow[kc + 16 + offA + j];
    v16bf a_hi, a_lo;
#pragma unroll
    for (int j = 0; j < 16; ++j) {
      __bf16 h = (__bf16)af[j];
      a_hi[j] = h;
      a_lo[j] = (__bf16)(af[j] - (float)h);
    }

    // ---- B fragment: 32x16 bf16 (N = n0+l16; K = kc + half*8 + (j&7) + (j>>3)*16) ----
    const int kb = kc + half * 8;
    const int nn = n0 + l16;
    v16bf b_hi, b_lo;
#pragma unroll
    for (int j = 0; j < 16; ++j) {
      float w  = W[(size_t)(kb + (j & 7) + (j >> 3) * 16) * 64 + nn];
      __bf16 h = (__bf16)w;
      b_hi[j] = h;
      b_lo[j] = (__bf16)(w - (float)h);
    }

    acc = __builtin_amdgcn_wmma_f32_16x16x32_bf16(false, a_hi, false, b_hi,
                                                  (short)0, acc, false, false);
    acc = __builtin_amdgcn_wmma_f32_16x16x32_bf16(false, a_hi, false, b_lo,
                                                  (short)0, acc, false, false);
    acc = __builtin_amdgcn_wmma_f32_16x16x32_bf16(false, a_lo, false, b_hi,
                                                  (short)0, acc, false, false);
  }

  // ---- epilogue: D layout is M = half*8 + i, N = n0 + l16 ----
  const int nn = n0 + l16;
  const float bv = bias ? bias[nn] : 0.0f;
#pragma unroll
  for (int i = 0; i < 8; ++i) {
    const int m = m0 + half * 8 + i;
    if (m < N) Y[(size_t)m * 64 + nn] = acc[i] + bv;
  }
}

// ---------------------------------------------------------------------------
// Segment-softmax scratch init: num=0, denom=0, emax=-inf
// ---------------------------------------------------------------------------
__global__ void __launch_bounds__(256)
init_segments(float* __restrict__ emax, float* __restrict__ denom,
              float* __restrict__ num, int n_dst)
{
  const int i = blockIdx.x * blockDim.x + threadIdx.x;
  if (i < n_dst * 64) num[i] = 0.0f;
  if (i < n_dst) { emax[i] = -INFINITY; denom[i] = 0.0f; }
}

// ---------------------------------------------------------------------------
// Pass 1: e = attn . leaky_relu(hs[src]+hd[dst]); atomic segment max.
// 16 lanes per edge, float4 per lane (coalesced 256B gathers).
// ---------------------------------------------------------------------------
__global__ void __launch_bounds__(256)
edge_pass1(const float* __restrict__ hs, const float* __restrict__ hd,
           const int* __restrict__ src, const int* __restrict__ dst,
           const float* __restrict__ attn,
           float* __restrict__ e_out, float* __restrict__ emax, int E)
{
  const int g = blockIdx.x * 16 + (threadIdx.x >> 4);  // edge id
  const int l = threadIdx.x & 15;
  if (g >= E) return;                                  // uniform per 16-group

  const int s = src[g];
  const int d = dst[g];
  const float4 a4 = ((const float4*)attn)[l];
  const float4 x  = ((const float4*)(hs + (size_t)s * 64))[l];
  const float4 y  = ((const float4*)(hd + (size_t)d * 64))[l];

  float t0 = x.x + y.x; t0 = (t0 > 0.0f) ? t0 : NEG_SLOPE * t0;
  float t1 = x.y + y.y; t1 = (t1 > 0.0f) ? t1 : NEG_SLOPE * t1;
  float t2 = x.z + y.z; t2 = (t2 > 0.0f) ? t2 : NEG_SLOPE * t2;
  float t3 = x.w + y.w; t3 = (t3 > 0.0f) ? t3 : NEG_SLOPE * t3;
  float sum = a4.x * t0 + a4.y * t1 + a4.z * t2 + a4.w * t3;

#pragma unroll
  for (int off = 8; off >= 1; off >>= 1)  // xor-reduce inside 16-lane group
    sum += __shfl_xor(sum, off, 32);

  if (l == 0) {
    e_out[g] = sum;
    atomic_max_f32(&emax[d], sum);
  }
}

// ---------------------------------------------------------------------------
// Pass 2: a = exp(e - emax[dst]); atomic add denom and a*hs[src] message.
// ---------------------------------------------------------------------------
__global__ void __launch_bounds__(256)
edge_pass2(const float* __restrict__ hs, const float* __restrict__ e_in,
           const int* __restrict__ src, const int* __restrict__ dst,
           const float* __restrict__ emax,
           float* __restrict__ denom, float* __restrict__ num, int E)
{
  const int g = blockIdx.x * 16 + (threadIdx.x >> 4);
  const int l = threadIdx.x & 15;
  if (g >= E) return;

  const int s = src[g];
  const int d = dst[g];
  const float a = __expf(e_in[g] - emax[d]);

  const float4 x = ((const float4*)(hs + (size_t)s * 64))[l];
  float* np = num + (size_t)d * 64 + l * 4;
  atomicAdd(np + 0, x.x * a);
  atomicAdd(np + 1, x.y * a);
  atomicAdd(np + 2, x.z * a);
  atomicAdd(np + 3, x.w * a);
  if (l == 0) atomicAdd(&denom[d], a);
}

// ---------------------------------------------------------------------------
// Pass 3: out[v, :] = num[v, :] / max(denom[v], 1e-9) + bias
// ---------------------------------------------------------------------------
__global__ void __launch_bounds__(256)
finalize_seg(const float* __restrict__ num, const float* __restrict__ denom,
             const float* __restrict__ bias, float* __restrict__ out,
             int n_dst, int ld_out)
{
  const int i = blockIdx.x * blockDim.x + threadIdx.x;
  if (i >= n_dst * 64) return;
  const int v = i >> 6;
  const int dcol = i & 63;
  const float den = fmaxf(denom[v], 1e-9f);
  out[(size_t)v * ld_out + dcol] = num[i] / den + bias[dcol];
}

// ---------------------------------------------------------------------------
extern "C" void kernel_launch(void* const* d_in, const int* in_sizes, int n_in,
                              void* d_out, int out_size, void* d_ws, size_t ws_size,
                              hipStream_t stream)
{
  const float* h_user   = (const float*)d_in[0];
  const float* h_item   = (const float*)d_in[1];
  const int*   rate_src = (const int*)d_in[2];
  const int*   rate_dst = (const int*)d_in[3];
  const int*   rb_src   = (const int*)d_in[4];
  const int*   rb_dst   = (const int*)d_in[5];
  const int*   link_src = (const int*)d_in[6];
  const int*   link_dst = (const int*)d_in[7];

  const float* w_src_r1  = (const float*)d_in[8];
  const float* w_dst_r1  = (const float*)d_in[9];
  const float* a_r1      = (const float*)d_in[10];
  const float* b_r1      = (const float*)d_in[11];
  const float* w_src_rb1 = (const float*)d_in[12];
  const float* w_dst_rb1 = (const float*)d_in[13];
  const float* a_rb1     = (const float*)d_in[14];
  const float* b_rb1     = (const float*)d_in[15];
  const float* w_src_rb2 = (const float*)d_in[16];
  const float* w_dst_rb2 = (const float*)d_in[17];
  const float* a_rb2     = (const float*)d_in[18];
  const float* b_rb2     = (const float*)d_in[19];
  const float* w_src_l2  = (const float*)d_in[20];
  const float* w_dst_l2  = (const float*)d_in[21];
  const float* a_l2      = (const float*)d_in[22];
  const float* b_l2      = (const float*)d_in[23];
  const float* w_out     = (const float*)d_in[24];
  const float* b_out     = (const float*)d_in[25];

  const int N_USER = in_sizes[0] / 64;
  const int N_ITEM = in_sizes[1] / 64;
  const int E_UI   = in_sizes[2];
  const int E_UU   = in_sizes[6];

  // ---- workspace layout (floats) ----
  float* ws = (float*)d_ws;
  const size_t NI64 = (size_t)N_ITEM * 64;
  const size_t NU64 = (size_t)N_USER * 64;
  const size_t Emx  = (size_t)((E_UI > E_UU) ? E_UI : E_UU);
  float* hs      = ws;  ws += NI64;
  float* hd      = ws;  ws += NI64;
  float* num     = ws;  ws += NI64;
  float* denom   = ws;  ws += N_ITEM;
  float* emax    = ws;  ws += N_ITEM;
  float* ebuf    = ws;  ws += Emx;
  float* h1_item = ws;  ws += NI64;
  float* h2_user = ws;  ws += NU64;
  float* cat     = ws;  ws += (size_t)N_USER * 128;

  auto run_gat = [&](const float* x_src, int n_src, const float* x_dst, int n_dst,
                     const int* src, const int* dst, int E,
                     const float* wsrc, const float* wdst,
                     const float* attn, const float* bias,
                     float* out, int ld_out) {
    gemm_wmma_bf16x3<<<(n_src + 31) / 32, 256, 0, stream>>>(x_src, 64, 64, wsrc, nullptr, hs, n_src);
    gemm_wmma_bf16x3<<<(n_dst + 31) / 32, 256, 0, stream>>>(x_dst, 64, 64, wdst, nullptr, hd, n_dst);
    init_segments<<<(n_dst * 64 + 255) / 256, 256, 0, stream>>>(emax, denom, num, n_dst);
    edge_pass1<<<(E + 15) / 16, 256, 0, stream>>>(hs, hd, src, dst, attn, ebuf, emax, E);
    edge_pass2<<<(E + 15) / 16, 256, 0, stream>>>(hs, ebuf, src, dst, emax, denom, num, E);
    finalize_seg<<<(n_dst * 64 + 255) / 256, 256, 0, stream>>>(num, denom, bias, out, n_dst, ld_out);
  };

  // layer1 'rate':      user -> item
  run_gat(h_user,  N_USER, h_item, N_ITEM, rate_src, rate_dst, E_UI,
          w_src_r1,  w_dst_r1,  a_r1,  b_r1,  h1_item, 64);
  // layer1 'rated-by':  item -> user
  run_gat(h_item,  N_ITEM, h_user, N_USER, rb_src, rb_dst, E_UI,
          w_src_rb1, w_dst_rb1, a_rb1, b_rb1, h2_user, 64);
  // layer2 'rated-by':  h1_item -> user   (writes cat[:, 0:64])
  run_gat(h1_item, N_ITEM, h_user, N_USER, rb_src, rb_dst, E_UI,
          w_src_rb2, w_dst_rb2, a_rb2, b_rb2, cat, 128);
  // layer2 'link':      h2_user -> user   (writes cat[:, 64:128])
  run_gat(h2_user, N_USER, h_user, N_USER, link_src, link_dst, E_UU,
          w_src_l2,  w_dst_l2,  a_l2,  b_l2,  cat + 64, 128);

  // out = cat @ w_out + b_out
  gemm_wmma_bf16x3<<<(N_USER + 31) / 32, 256, 0, stream>>>(
      cat, 128, 128, w_out, b_out, (float*)d_out, N_USER);
}
